// Model_random_proto_44315472560255
// MI455X (gfx1250) — matrix operand order repaired and checked by reading, not compiled, not run on hardware
//
#include <hip/hip_runtime.h>
#include <hip/hip_bf16.h>
#include <math.h>

#define HID   128
#define FINF  256
#define CNUM  40

typedef __attribute__((ext_vector_type(16))) __bf16         v16bf;
typedef __attribute__((ext_vector_type(8)))  unsigned short v8u;
typedef __attribute__((ext_vector_type(8)))  float          v8f;

struct u16x16 { v8u lo; v8u hi; };   // 32 bytes -> bit_cast to v16bf

static __device__ __forceinline__ unsigned short f2bf_bits(float f) {
    union { float f; unsigned u; } v; v.f = f;
    unsigned u = v.u;
    unsigned r = u + 0x7fffu + ((u >> 16) & 1u);   // round-to-nearest-even
    return (unsigned short)(r >> 16);
}

// ---------------------------------------------------------------------------
// GCN normalization: deg = 1 (self loop) + #incoming edges; dinv = rsqrt(deg)
// ---------------------------------------------------------------------------
__global__ void deg_init_kernel(float* __restrict__ deg, int N) {
    int i = blockIdx.x * blockDim.x + threadIdx.x;
    if (i < N) deg[i] = 1.0f;
}
__global__ void deg_count_kernel(float* __restrict__ deg, const int* __restrict__ ei, int E) {
    int i = blockIdx.x * blockDim.x + threadIdx.x;
    if (i < E) atomicAdd(&deg[ei[(size_t)E + i]], 1.0f);
}
__global__ void deg_rsqrt_kernel(float* __restrict__ d, int N) {
    int i = blockIdx.x * blockDim.x + threadIdx.x;
    if (i < N) d[i] = rsqrtf(d[i]);
}

// ---------------------------------------------------------------------------
// f32 -> bf16 conversion passes (optional fused ReLU)
// ---------------------------------------------------------------------------
template<int RELU>
__global__ void cvt_bf16_kernel(const float* __restrict__ src,
                                unsigned short* __restrict__ dst, int total)
{
    int i = blockIdx.x * blockDim.x + threadIdx.x;
    if (i >= total) return;
    float f = src[i];
    if (RELU) f = fmaxf(f, 0.0f);
    dst[i] = f2bf_bits(f);
}

// W (K x N, f32, row-major) -> Wt (N x K, bf16)
__global__ void cvt_weight_T_kernel(const float* __restrict__ W,
                                    unsigned short* __restrict__ Wt, int K, int N)
{
    int i = blockIdx.x * blockDim.x + threadIdx.x;
    if (i >= K * N) return;
    int k = i / N, n = i - k * N;
    Wt[(size_t)n * K + k] = f2bf_bits(W[i]);
}

// ---------------------------------------------------------------------------
// bf16 WMMA GEMM:  D[M x 128] = A[M x K] * Bt[128 x K]^T
//   A  : bf16 row-major M x K
//   Bt : bf16 row-major 128 x K   (i.e. B stored column-major)
//   EPI: 0 -> store acc ; 1 -> store (acc+1)*0.5
// One wave per 16x16 tile; 8 waves/block cover the 128 output columns;
// fully unrolled K loop (K is a template parameter: 256 or 128).
// Fragment layouts per CDNA5 ISA 7.12.2 (wave32); all loads are 16B b128.
// ---------------------------------------------------------------------------
template<int K, int EPI>
__global__ __launch_bounds__(256)
void wmma_gemm_bf16_kernel(const unsigned short* __restrict__ A,
                           const unsigned short* __restrict__ Bt,
                           float* __restrict__ D)
{
    const int lane = threadIdx.x & 31;
    const int wave = threadIdx.x >> 5;
    const int rowT = blockIdx.x;            // 16-row tile
    const int colT = wave;                  // 16-col tile (8 waves = 128 cols)

    const int m  = lane & 15;
    const int kh = (lane >> 4) << 3;        // A: lane-half K offset (0 / 8)
    const int ks = (lane >> 4) << 3;        // B: lane-half K offset (0 / 8)
    const int n  = colT * 16 + (lane & 15);

    const unsigned short* ap = A  + (size_t)(rowT * 16 + m) * K + kh;
    const unsigned short* bp = Bt + (size_t)n * K + ks;

    v8f acc = {};
    #pragma unroll
    for (int kb = 0; kb < K; kb += 32) {
        u16x16 a, b;
        a.lo = *(const v8u*)(ap + kb);        // K = kb+kh+0..7
        a.hi = *(const v8u*)(ap + kb + 16);   // K = kb+kh+16..23
        b.lo = *(const v8u*)(bp + kb);        // K = kb+ks+0..7
        b.hi = *(const v8u*)(bp + kb + 16);   // K = kb+ks+16..23
        acc = __builtin_amdgcn_wmma_f32_16x16x32_bf16(
                  false, __builtin_bit_cast(v16bf, a),
                  false, __builtin_bit_cast(v16bf, b),
                  (short)0, acc, false, false);
    }

    // C/D layout: VGPR v -> row M = v + 8*(lane>=16); col N = lane&15
    const int mb = rowT * 16 + ((lane >> 4) << 3);
    #pragma unroll
    for (int v = 0; v < 8; ++v) {
        float r = acc[v];
        if (EPI == 1) r = (r + 1.0f) * 0.5f;
        D[(size_t)(mb + v) * HID + n] = r;
    }
}

// ---------------------------------------------------------------------------
// Aggregation: out = dinv[i]^2 * t[i] + bias  (self-loop), then edge atomics
// ---------------------------------------------------------------------------
__global__ void agg_init_kernel(const float* __restrict__ t, float* __restrict__ out,
                                const float* __restrict__ dinv,
                                const float* __restrict__ bias, int N)
{
    int i = blockIdx.x * blockDim.x + threadIdx.x;
    if (i >= N * HID) return;
    int row = i >> 7, col = i & (HID - 1);
    float dv = dinv[row];
    out[i] = dv * dv * t[i] + bias[col];
}

// one wave per edge; each lane handles 4 of the 128 features (float4 read)
__global__ void agg_edge_kernel(const float* __restrict__ t, float* __restrict__ out,
                                const int* __restrict__ ei,
                                const float* __restrict__ dinv, int E)
{
    int wave = threadIdx.x >> 5;
    int lane = threadIdx.x & 31;
    int e = blockIdx.x * 8 + wave;
    if (e >= E) return;
    int s = ei[e];
    int d = ei[(size_t)E + e];
    float w = dinv[s] * dinv[d];
    const float4 v = ((const float4*)(t + (size_t)s * HID))[lane];
    float* od = out + (size_t)d * HID + lane * 4;
    atomicAdd(od + 0, w * v.x);
    atomicAdd(od + 1, w * v.y);
    atomicAdd(od + 2, w * v.z);
    atomicAdd(od + 3, w * v.w);
}

// ---------------------------------------------------------------------------
// In-place L2 row normalization (one wave per row)
// ---------------------------------------------------------------------------
__global__ void l2norm_kernel(float* __restrict__ h, int N)
{
    int wave = threadIdx.x >> 5, lane = threadIdx.x & 31;
    int row = blockIdx.x * 8 + wave;
    if (row >= N) return;
    float4* p = (float4*)(h + (size_t)row * HID);
    float4 v = p[lane];
    float s = v.x * v.x + v.y * v.y + v.z * v.z + v.w * v.w;
    #pragma unroll
    for (int off = 16; off > 0; off >>= 1) s += __shfl_xor(s, off, 32);
    float inv = rsqrtf(s);
    v.x *= inv; v.y *= inv; v.z *= inv; v.w *= inv;
    p[lane] = v;
}

// anchors: f32 copy (prototype head) + bf16 copy (WMMA B operand, N x K layout)
__global__ void gather_anchor_kernel(const float* __restrict__ h,
                                     const int* __restrict__ prot,
                                     float* __restrict__ anchorsF,
                                     unsigned short* __restrict__ anchorsBf)
{
    int p = blockIdx.x, j = threadIdx.x;
    float f = h[(size_t)prot[p] * HID + j];
    anchorsF [(size_t)p * HID + j] = f;
    anchorsBf[(size_t)p * HID + j] = f2bf_bits(f);
}

// ---------------------------------------------------------------------------
// Prototype head: hp = relu(anchors@lw1+lb1); out_proto = log_softmax(hp@lw2+lb2)
// ---------------------------------------------------------------------------
__global__ void proto_head_kernel(const float* __restrict__ anchors,
                                  const float* __restrict__ lw1, const float* __restrict__ lb1,
                                  const float* __restrict__ lw2, const float* __restrict__ lb2,
                                  float* __restrict__ out_proto)
{
    __shared__ float sa[HID];
    __shared__ float shp[HID];
    __shared__ float slog[CNUM];
    __shared__ float sred[2];
    int p = blockIdx.x, j = threadIdx.x;
    sa[j] = anchors[(size_t)p * HID + j];
    __syncthreads();
    float acc = lb1[j];
    for (int k = 0; k < HID; ++k) acc += sa[k] * lw1[(size_t)k * HID + j];
    shp[j] = fmaxf(acc, 0.0f);
    __syncthreads();
    if (j < CNUM) {
        float a2 = lb2[j];
        for (int k = 0; k < HID; ++k) a2 += shp[k] * lw2[(size_t)k * CNUM + j];
        slog[j] = a2;
    }
    __syncthreads();
    if (j == 0) {
        float m = -INFINITY;
        for (int c = 0; c < CNUM; ++c) m = fmaxf(m, slog[c]);
        float s = 0.0f;
        for (int c = 0; c < CNUM; ++c) s += expf(slog[c] - m);
        sred[0] = m; sred[1] = logf(s);
    }
    __syncthreads();
    if (j < CNUM) out_proto[(size_t)p * CNUM + j] = slog[j] - sred[0] - sred[1];
}

// ---------------------------------------------------------------------------
// Final: out = log_softmax(x_rel @ out_proto)  (one 64-thread block per row)
// ---------------------------------------------------------------------------
__global__ void out_final_kernel(const float* __restrict__ x_rel,
                                 const float* __restrict__ out_proto,
                                 float* __restrict__ out, int P)
{
    __shared__ float sx[HID];
    __shared__ float slog[CNUM];
    __shared__ float sred[2];
    int row = blockIdx.x, j = threadIdx.x;
    sx[j]      = x_rel[(size_t)row * P + j];
    sx[j + 64] = x_rel[(size_t)row * P + j + 64];
    __syncthreads();
    if (j < CNUM) {
        float a = 0.0f;
        for (int k = 0; k < P; ++k) a += sx[k] * out_proto[(size_t)k * CNUM + j];
        slog[j] = a;
    }
    __syncthreads();
    if (j == 0) {
        float m = -INFINITY;
        for (int c = 0; c < CNUM; ++c) m = fmaxf(m, slog[c]);
        float s = 0.0f;
        for (int c = 0; c < CNUM; ++c) s += expf(slog[c] - m);
        sred[0] = m; sred[1] = logf(s);
    }
    __syncthreads();
    if (j < CNUM) out[(size_t)row * CNUM + j] = slog[j] - sred[0] - sred[1];
}

// ---------------------------------------------------------------------------
extern "C" void kernel_launch(void* const* d_in, const int* in_sizes, int n_in,
                              void* d_out, int out_size, void* d_ws, size_t ws_size,
                              hipStream_t stream)
{
    const float* x    = (const float*)d_in[0];
    const int*   ei   = (const int*)  d_in[1];
    const int*   prot = (const int*)  d_in[2];
    const float* W0   = (const float*)d_in[3];
    const float* b0   = (const float*)d_in[4];
    const float* Wh   = (const float*)d_in[5];
    const float* bh   = (const float*)d_in[6];
    const float* lw1  = (const float*)d_in[7];
    const float* lb1  = (const float*)d_in[8];
    const float* lw2  = (const float*)d_in[9];
    const float* lb2  = (const float*)d_in[10];

    const int N = in_sizes[0] / FINF;   // 50000 (multiple of 16)
    const int E = in_sizes[1] / 2;      // 1.6M
    const int P = in_sizes[2];          // 128

    // ---- workspace layout ----
    float* ws      = (float*)d_ws;
    float* dinv    = ws;                               // N f32
    float* bufA    = dinv + N;                         // N*HID f32 (pre-agg)
    float* bufB    = bufA + (size_t)N * HID;           // N*HID f32 (aggregated h)
    float* anchF   = bufB + (size_t)N * HID;           // P*HID f32
    unsigned short* W0t   = (unsigned short*)(anchF + (size_t)P * HID); // HID*FINF
    unsigned short* Wh0t  = W0t  + (size_t)HID * FINF;                  // HID*HID
    unsigned short* Wh1t  = Wh0t + (size_t)HID * HID;                   // HID*HID
    unsigned short* anchB = Wh1t + (size_t)HID * HID;                   // P*HID bf16
    unsigned short* abf   = anchB + (size_t)P * HID;                    // N*FINF bf16 (reused)

    float* out       = (float*)d_out;                  // N*CNUM
    float* x_rel     = out + (size_t)N * CNUM;         // N*P
    float* out_proto = x_rel + (size_t)N * P;          // P*CNUM

    const int rowTiles = N / 16;

    // GCN normalization coefficients
    deg_init_kernel <<<(N + 255) / 256, 256, 0, stream>>>(dinv, N);
    deg_count_kernel<<<(E + 255) / 256, 256, 0, stream>>>(dinv, ei, E);
    deg_rsqrt_kernel<<<(N + 255) / 256, 256, 0, stream>>>(dinv, N);

    // weights: convert + transpose to bf16 N x K once
    cvt_weight_T_kernel<<<(FINF * HID + 255) / 256, 256, 0, stream>>>(W0, W0t, FINF, HID);
    cvt_weight_T_kernel<<<(HID * HID + 255) / 256, 256, 0, stream>>>(Wh, Wh0t, HID, HID);
    cvt_weight_T_kernel<<<(HID * HID + 255) / 256, 256, 0, stream>>>(Wh + (size_t)HID * HID,
                                                                     Wh1t, HID, HID);

    const int aggBlocks  = (N * HID + 255) / 256;
    const int edgeBlocks = (E + 7) / 8;
    const int cvtBigBlk  = (N * FINF + 255) / 256;
    const int cvtHidBlk  = (N * HID + 255) / 256;

    // Layer 1: t = x @ W0 ; h = agg(t) + b0
    cvt_bf16_kernel<0><<<cvtBigBlk, 256, 0, stream>>>(x, abf, N * FINF);
    wmma_gemm_bf16_kernel<FINF, 0><<<rowTiles, 256, 0, stream>>>(abf, W0t, bufA);
    agg_init_kernel<<<aggBlocks, 256, 0, stream>>>(bufA, bufB, dinv, b0, N);
    agg_edge_kernel<<<edgeBlocks, 256, 0, stream>>>(bufA, bufB, ei, dinv, E);

    // Layer 2: t = relu(h) @ Wh[0] ; h = agg(t) + bh[0]
    cvt_bf16_kernel<1><<<cvtHidBlk, 256, 0, stream>>>(bufB, abf, N * HID);
    wmma_gemm_bf16_kernel<HID, 0><<<rowTiles, 256, 0, stream>>>(abf, Wh0t, bufA);
    agg_init_kernel<<<aggBlocks, 256, 0, stream>>>(bufA, bufB, dinv, bh, N);
    agg_edge_kernel<<<edgeBlocks, 256, 0, stream>>>(bufA, bufB, ei, dinv, E);

    // Layer 3: t = relu(h) @ Wh[1] ; h = agg(t) + bh[1]
    cvt_bf16_kernel<1><<<cvtHidBlk, 256, 0, stream>>>(bufB, abf, N * HID);
    wmma_gemm_bf16_kernel<HID, 0><<<rowTiles, 256, 0, stream>>>(abf, Wh1t, bufA);
    agg_init_kernel<<<aggBlocks, 256, 0, stream>>>(bufA, bufB, dinv, bh + HID, N);
    agg_edge_kernel<<<edgeBlocks, 256, 0, stream>>>(bufA, bufB, ei, dinv, E);

    // L2 normalize rows in place, gather anchors (f32 + bf16)
    l2norm_kernel<<<(N + 7) / 8, 256, 0, stream>>>(bufB, N);
    gather_anchor_kernel<<<P, HID, 0, stream>>>(bufB, prot, anchF, anchB);

    // x_rel = (hnorm @ anchors^T + 1) * 0.5  (anchors already N x K bf16)
    cvt_bf16_kernel<0><<<cvtHidBlk, 256, 0, stream>>>(bufB, abf, N * HID);
    wmma_gemm_bf16_kernel<HID, 1><<<rowTiles, 256, 0, stream>>>(abf, anchB, x_rel);

    // prototype head and final log-softmax
    proto_head_kernel<<<P, HID, 0, stream>>>(anchF, lw1, lb1, lw2, lb2, out_proto);
    out_final_kernel<<<N, 64, 0, stream>>>(x_rel, out_proto, out, P);
}
